// ModernBertAttention_15393162789338
// MI455X (gfx1250) — matrix operand order repaired and checked by reading, not compiled, not run on hardware
//
#include <hip/hip_runtime.h>

#define T_TOK 4096
#define D_MODEL 768
#define NH 12
#define DHD 64
#define SEGL 512
#define N_QKV 2304

typedef __bf16 bf16_t;
typedef __attribute__((ext_vector_type(16))) __bf16 v16bf;
typedef __attribute__((ext_vector_type(8)))  __bf16 v8bf;
typedef __attribute__((ext_vector_type(8)))  float  v8f;
typedef __attribute__((ext_vector_type(4)))  unsigned int v4u;
typedef __attribute__((ext_vector_type(8)))  int v8i;
typedef __attribute__((ext_vector_type(4)))  int v4i;

#if __has_builtin(__builtin_amdgcn_tensor_load_to_lds) && \
    __has_builtin(__builtin_amdgcn_s_wait_tensorcnt)
#define HAVE_TDM 1
#else
#define HAVE_TDM 0
#endif

// ---------- helpers ----------

__device__ __forceinline__ bf16_t f2bf(float f) {
  unsigned u = __float_as_uint(f);
  unsigned r = (u + 0x7FFFu + ((u >> 16) & 1u)) >> 16;
  unsigned short us = (unsigned short)r;
  return __builtin_bit_cast(bf16_t, us);
}

__device__ __forceinline__ v8f zero8() {
  v8f z;
#pragma unroll
  for (int i = 0; i < 8; ++i) z[i] = 0.0f;
  return z;
}

// flat LDS aperture: addr[31:0] == LDS byte offset (ISA aperture rule)
__device__ __forceinline__ unsigned lds_off(const void* p) {
  return (unsigned)(size_t)p;
}

#if HAVE_TDM
// TDM: load a 2D tile (tile_k contiguous bf16 elems x tile_n rows, row stride
// ld_elems) from global into LDS at ldsDst. Builds D# group0/group1 per ISA
// CDNA5 ISA; groups 2/3 zero (2D tensor). 6-arg builtin form
// (clang-23 / therock-10.0 headers).
__device__ __forceinline__ void tdm_load_2d(const bf16_t* gsrc, unsigned ldsDst,
                                            int tile_k, int tile_n, int ld_elems,
                                            int tensor_k, int tensor_n) {
  unsigned long long ga = (unsigned long long)(size_t)gsrc;
  v4u g0;
  g0[0] = 1u;                                         // count=1 (valid user D#)
  g0[1] = ldsDst;                                     // lds_addr
  g0[2] = (unsigned)(ga & 0xFFFFFFFFu);               // global_addr[31:0]
  g0[3] = (unsigned)((ga >> 32) & 0x1FFFFFFu) | (2u << 30);  // addr[56:32] | type=2
  v8i g1;
  g1[0] = (int)(1u << 16);                            // data_size=1 (2 bytes)
  g1[1] = (int)(((unsigned)tensor_k & 0xFFFFu) << 16);          // tensor_dim0 lo
  g1[2] = (int)((((unsigned)tensor_k >> 16) & 0xFFFFu) |
                (((unsigned)tensor_n & 0xFFFFu) << 16));        // dim0 hi | dim1 lo
  g1[3] = (int)((((unsigned)tensor_n >> 16) & 0xFFFFu) |
                (((unsigned)tile_k & 0xFFFFu) << 16));          // dim1 hi | tile_dim0
  g1[4] = (int)((unsigned)tile_n & 0xFFFFu);          // tile_dim1 (tile_dim2=0)
  g1[5] = ld_elems;                                   // tensor_dim0_stride[31:0]
  g1[6] = 0;
  g1[7] = 0;
  v4i z4 = {0, 0, 0, 0};
  v8i z8;
#pragma unroll
  for (int i = 0; i < 8; ++i) z8[i] = 0;
  __builtin_amdgcn_tensor_load_to_lds(g0, g1, z4, z4, z8, 0);
}
#endif

// 16x32 bf16 A-tile, contiguous-k order (same as load_b_tile): lane L holds row M=L&15,
// K elems kb..kb+15 contiguous, kb=(L>=16)*16.
__device__ __forceinline__ v16bf load_a_tile(const bf16_t* X, int ld) {
  int lane = threadIdx.x & 31;
  int m  = lane & 15;
  int kb = (lane >> 4) << 4;
  v8bf lo = *(const v8bf*)(X + (size_t)m * ld + kb);
  v8bf hi = *(const v8bf*)(X + (size_t)m * ld + kb + 8);
  v16bf a;
#pragma unroll
  for (int i = 0; i < 8; ++i) { a[i] = lo[i]; a[8 + i] = hi[i]; }
  return a;
}

// 32x16 bf16 B-tile from [N][K]-major: lane L holds col N=L&15,
// K elems kb..kb+15 contiguous, kb=(L>=16)*16.
__device__ __forceinline__ v16bf load_b_tile(const bf16_t* Wt, int ld) {
  int lane = threadIdx.x & 31;
  int n  = lane & 15;
  int kb = (lane >> 4) << 4;
  return *(const v16bf*)(Wt + (size_t)n * ld + kb);
}

__device__ __forceinline__ v8f wmma_bf16(v16bf a, v16bf b, v8f c) {
  return __builtin_amdgcn_wmma_f32_16x16x32_bf16(false, a, false, b,
                                                 (short)0, c, false, false);
}

// ---------- kernel 0: fp32 -> bf16 conversions ----------

__global__ void cvt_f32_bf16(const float* __restrict__ src,
                             bf16_t* __restrict__ dst, int n) {
  int i = blockIdx.x * blockDim.x + threadIdx.x;
  if (i < n) dst[i] = f2bf(src[i]);
}

__global__ void cvt_transpose(const float* __restrict__ src,
                              bf16_t* __restrict__ dst, int R, int C) {
  int i = blockIdx.x * blockDim.x + threadIdx.x;
  if (i < R * C) {
    int r = i / C, c = i % C;
    dst[(size_t)c * R + r] = f2bf(src[i]);
  }
}

// ---------- kernel 1: QKV GEMM (TDM-staged weights) + NeoX RoPE ----------

__global__ void __launch_bounds__(256) qkv_rope_kernel(
    const bf16_t* __restrict__ X, const bf16_t* __restrict__ Wt,
    const int* __restrict__ pos_ids,
    bf16_t* __restrict__ Qb, bf16_t* __restrict__ Kb, bf16_t* __restrict__ Vt) {
  __shared__ __align__(128) bf16_t sW[2][256 * 32];  // 32 KB double buffer

  const int lane = threadIdx.x & 31;
  const int wid  = threadIdx.x >> 5;
  const int tid  = threadIdx.x;
  const int colWG = (blockIdx.x % 9) * 256;   // 2304/256 = 9
  const int rowWG = (blockIdx.x / 9) * 64;
  const int waveCol = colWG + (wid & 3) * 64;
  const int waveRow = rowWG + (wid >> 2) * 32;
  const int localN  = (wid & 3) * 64;

  v8f acc[2][4];
#pragma unroll
  for (int mi = 0; mi < 2; ++mi)
#pragma unroll
    for (int ni = 0; ni < 4; ++ni) acc[mi][ni] = zero8();

  // prologue: stage first 256(N) x 32(K) weight slab into sW[0]
#if HAVE_TDM
  if (wid == 0)
    tdm_load_2d(Wt + (size_t)colWG * D_MODEL, lds_off(&sW[0][0]),
                32, 256, D_MODEL, D_MODEL, N_QKV);
#else
  {
    const bf16_t* src = Wt + (size_t)(colWG + tid) * D_MODEL;
#pragma unroll
    for (int j = 0; j < 32; j += 8)
      *(v8bf*)&sW[0][tid * 32 + j] = *(const v8bf*)(src + j);
  }
#endif

  for (int kk = 0; kk < D_MODEL; kk += 32) {
    const int ib = (kk >> 5) & 1;
#if HAVE_TDM
    if (wid == 0) __builtin_amdgcn_s_wait_tensorcnt(0);
#endif
    __syncthreads();
    if (kk + 32 < D_MODEL) {
#if HAVE_TDM
      if (wid == 0)
        tdm_load_2d(Wt + (size_t)colWG * D_MODEL + kk + 32,
                    lds_off(&sW[ib ^ 1][0]), 32, 256, D_MODEL, D_MODEL, N_QKV);
#else
      const bf16_t* src = Wt + (size_t)(colWG + tid) * D_MODEL + kk + 32;
#pragma unroll
      for (int j = 0; j < 32; j += 8)
        *(v8bf*)&sW[ib ^ 1][tid * 32 + j] = *(const v8bf*)(src + j);
#endif
    }
    __builtin_prefetch(X + (size_t)waveRow * D_MODEL + kk + 128, 0, 0);
    v16bf a0 = load_a_tile(X + (size_t)waveRow * D_MODEL + kk, D_MODEL);
    v16bf a1 = load_a_tile(X + (size_t)(waveRow + 16) * D_MODEL + kk, D_MODEL);
    const bf16_t* sWb = &sW[ib][0];
#pragma unroll
    for (int ni = 0; ni < 4; ++ni) {
      v16bf b = load_b_tile(sWb + (size_t)(localN + ni * 16) * 32, 32);
      acc[0][ni] = wmma_bf16(a0, b, acc[0][ni]);
      acc[1][ni] = wmma_bf16(a1, b, acc[1][ni]);
    }
  }

  const int kind = waveCol / D_MODEL;          // 0=Q, 1=K, 2=V
  const int h    = (waveCol % D_MODEL) >> 6;
  const int hi8  = (lane >> 4) << 3;
  const int nl   = lane & 15;

  if (kind < 2) {
    bf16_t* dst = (kind == 0) ? Qb : Kb;
#pragma unroll
    for (int mi = 0; mi < 2; ++mi) {
#pragma unroll
      for (int v = 0; v < 8; ++v) {
        int t = waveRow + mi * 16 + hi8 + v;
        float pos = (float)pos_ids[t];
#pragma unroll
        for (int ni = 0; ni < 2; ++ni) {      // rotate pairs (d, d+32)
          float d = (float)(ni * 16 + nl);
          float ang = pos * __expf(-0.37446653f * d);  // ln(160000)/32
          float s, c;
          __sincosf(ang, &s, &c);
          float x1 = acc[mi][ni][v], x2 = acc[mi][ni + 2][v];
          acc[mi][ni][v]     = x1 * c - x2 * s;
          acc[mi][ni + 2][v] = x2 * c + x1 * s;
        }
#pragma unroll
        for (int ni = 0; ni < 4; ++ni) {
          int dh = ni * 16 + nl;
          dst[((size_t)h * T_TOK + t) * DHD + dh] = f2bf(acc[mi][ni][v]);
        }
      }
    }
  } else {
#pragma unroll
    for (int mi = 0; mi < 2; ++mi)
#pragma unroll
      for (int v = 0; v < 8; ++v) {
        int t = waveRow + mi * 16 + hi8 + v;
#pragma unroll
        for (int ni = 0; ni < 4; ++ni) {
          int dh = ni * 16 + nl;
          Vt[((size_t)h * DHD + dh) * T_TOK + t] = f2bf(acc[mi][ni][v]);
        }
      }
  }
}

// ---------- kernel 2: segment-block flash attention (TDM-staged K/V) ----------
// grid = 12 heads * 8 segs * 4 q-chunks = 384 WGs; wave handles 16 q rows.

__global__ void __launch_bounds__(256) attn_kernel(
    const bf16_t* __restrict__ Qb, const bf16_t* __restrict__ Kb,
    const bf16_t* __restrict__ Vt, bf16_t* __restrict__ Ctx) {
  __shared__ __align__(128) bf16_t sK[2][32 * 64];   // 8 KB: 32 keys x 64 dh
  __shared__ __align__(128) bf16_t sV[2][64 * 32];   // 8 KB: 64 dh x 32 keys
  __shared__ __align__(32)  bf16_t sP[8][16 * 32];   // 8 KB: per-wave P buffer

  const int lane = threadIdx.x & 31;
  const int wid  = threadIdx.x >> 5;
  const int tid  = threadIdx.x;
  const int h    = blockIdx.x >> 5;
  const int rem  = blockIdx.x & 31;
  const int seg  = rem >> 2;
  const int qrow0 = seg * SEGL + (rem & 3) * 128 + wid * 16;
  const int kc0   = seg * SEGL;

  const bf16_t* Qh = Qb + (size_t)h * T_TOK * DHD;
  const bf16_t* Kh = Kb + (size_t)h * T_TOK * DHD;
  const bf16_t* Vh = Vt + (size_t)h * DHD * T_TOK;

  v16bf aq0 = load_a_tile(Qh + (size_t)qrow0 * DHD, DHD);       // dh 0..31
  v16bf aq1 = load_a_tile(Qh + (size_t)qrow0 * DHD + 32, DHD);  // dh 32..63

  v8f acc_o[4];
#pragma unroll
  for (int ni = 0; ni < 4; ++ni) acc_o[ni] = zero8();
  float rmax[8], rsum[8];
#pragma unroll
  for (int v = 0; v < 8; ++v) { rmax[v] = -1e30f; rsum[v] = 0.0f; }

  const float SC = 0.125f;  // 1/sqrt(64)
  const int nl  = lane & 15;
  const int hi8 = (lane >> 4) << 3;

  // prologue: stage first 32-key chunk of K and V
#if HAVE_TDM
  if (wid == 0) {
    tdm_load_2d(Kh + (size_t)kc0 * DHD, lds_off(&sK[0][0]), 64, 32, DHD, DHD, T_TOK);
    tdm_load_2d(Vh + kc0, lds_off(&sV[0][0]), 32, 64, T_TOK, T_TOK, DHD);
  }
#else
  *(v8bf*)&sK[0][tid * 8] = *(const v8bf*)(Kh + (size_t)kc0 * DHD + tid * 8);
  {
    int r = tid >> 2, c0 = (tid & 3) * 8;
    *(v8bf*)&sV[0][r * 32 + c0] = *(const v8bf*)(Vh + (size_t)r * T_TOK + kc0 + c0);
  }
#endif

  for (int ic = 0; ic < SEGL / 32; ++ic) {
    const int ib = ic & 1;
#if HAVE_TDM
    if (wid == 0) __builtin_amdgcn_s_wait_tensorcnt(0);
#endif
    __syncthreads();
    if (ic + 1 < SEGL / 32) {
      const int kn = kc0 + (ic + 1) * 32;
#if HAVE_TDM
      if (wid == 0) {
        tdm_load_2d(Kh + (size_t)kn * DHD, lds_off(&sK[ib ^ 1][0]),
                    64, 32, DHD, DHD, T_TOK);
        tdm_load_2d(Vh + kn, lds_off(&sV[ib ^ 1][0]),
                    32, 64, T_TOK, T_TOK, DHD);
      }
#else
      *(v8bf*)&sK[ib ^ 1][tid * 8] = *(const v8bf*)(Kh + (size_t)kn * DHD + tid * 8);
      {
        int r = tid >> 2, c0 = (tid & 3) * 8;
        *(v8bf*)&sV[ib ^ 1][r * 32 + c0] =
            *(const v8bf*)(Vh + (size_t)r * T_TOK + kn + c0);
      }
#endif
    }

    // scores from LDS-staged K (block fully in-segment -> no mask)
    const bf16_t* kbuf = &sK[ib][0];
    const bf16_t* vbuf = &sV[ib][0];
    v16bf b00 = load_b_tile(kbuf, 64);
    v16bf b01 = load_b_tile(kbuf + 32, 64);
    v16bf b10 = load_b_tile(kbuf + 16 * 64, 64);
    v16bf b11 = load_b_tile(kbuf + 16 * 64 + 32, 64);
    v8f s0 = wmma_bf16(aq0, b00, zero8());
    s0 = wmma_bf16(aq1, b01, s0);        // keys +0..15
    v8f s1 = wmma_bf16(aq0, b10, zero8());
    s1 = wmma_bf16(aq1, b11, s1);        // keys +16..31

    // online softmax in C layout (row m = v + hi8 across 16 lanes)
#pragma unroll
    for (int v = 0; v < 8; ++v) {
      float e0 = s0[v] * SC, e1 = s1[v] * SC;
      float mloc = fmaxf(e0, e1);
#pragma unroll
      for (int off = 1; off < 16; off <<= 1)
        mloc = fmaxf(mloc, __shfl_xor(mloc, off, 32));
      float nm = fmaxf(rmax[v], mloc);
      float cf = __expf(rmax[v] - nm);
      rmax[v] = nm;
      float p0 = __expf(e0 - nm);
      float p1 = __expf(e1 - nm);
      float ps = p0 + p1;
#pragma unroll
      for (int off = 1; off < 16; off <<= 1)
        ps += __shfl_xor(ps, off, 32);
      rsum[v] = rsum[v] * cf + ps;
#pragma unroll
      for (int ni = 0; ni < 4; ++ni) acc_o[ni][v] *= cf;
      int m = v + hi8;
      sP[wid][m * 32 + nl]      = f2bf(p0);
      sP[wid][m * 32 + 16 + nl] = f2bf(p1);
    }

    // re-load P as 16x32 A-tile (wave-private, DS in-order)
    v16bf pa = load_a_tile(&sP[wid][0], 32);
#pragma unroll
    for (int ni = 0; ni < 4; ++ni) {
      v16bf bv = load_b_tile(vbuf + (size_t)(ni * 16) * 32, 32);
      acc_o[ni] = wmma_bf16(pa, bv, acc_o[ni]);
    }
  }

#pragma unroll
  for (int v = 0; v < 8; ++v) {
    float inv = 1.0f / rsum[v];
    int t = qrow0 + v + hi8;
#pragma unroll
    for (int ni = 0; ni < 4; ++ni)
      Ctx[(size_t)t * D_MODEL + h * DHD + ni * 16 + nl] = f2bf(acc_o[ni][v] * inv);
  }
}

// ---------- kernel 3: output projection (TDM-staged Wo, fp32 result) ----------

__global__ void __launch_bounds__(256) out_proj_kernel(
    const bf16_t* __restrict__ Ctx, const bf16_t* __restrict__ WoT,
    float* __restrict__ Out) {
  __shared__ __align__(128) bf16_t sW[2][256 * 32];  // 32 KB double buffer

  const int lane = threadIdx.x & 31;
  const int wid  = threadIdx.x >> 5;
  const int tid  = threadIdx.x;
  const int colWG = (blockIdx.x % 3) * 256;  // 768/256 = 3
  const int rowWG = (blockIdx.x / 3) * 64;
  const int waveCol = colWG + (wid & 3) * 64;
  const int waveRow = rowWG + (wid >> 2) * 32;
  const int localN  = (wid & 3) * 64;

  v8f acc[2][4];
#pragma unroll
  for (int mi = 0; mi < 2; ++mi)
#pragma unroll
    for (int ni = 0; ni < 4; ++ni) acc[mi][ni] = zero8();

#if HAVE_TDM
  if (wid == 0)
    tdm_load_2d(WoT + (size_t)colWG * D_MODEL, lds_off(&sW[0][0]),
                32, 256, D_MODEL, D_MODEL, D_MODEL);
#else
  {
    const bf16_t* src = WoT + (size_t)(colWG + tid) * D_MODEL;
#pragma unroll
    for (int j = 0; j < 32; j += 8)
      *(v8bf*)&sW[0][tid * 32 + j] = *(const v8bf*)(src + j);
  }
#endif

  for (int kk = 0; kk < D_MODEL; kk += 32) {
    const int ib = (kk >> 5) & 1;
#if HAVE_TDM
    if (wid == 0) __builtin_amdgcn_s_wait_tensorcnt(0);
#endif
    __syncthreads();
    if (kk + 32 < D_MODEL) {
#if HAVE_TDM
      if (wid == 0)
        tdm_load_2d(WoT + (size_t)colWG * D_MODEL + kk + 32,
                    lds_off(&sW[ib ^ 1][0]), 32, 256, D_MODEL, D_MODEL, D_MODEL);
#else
      const bf16_t* src = WoT + (size_t)(colWG + tid) * D_MODEL + kk + 32;
#pragma unroll
      for (int j = 0; j < 32; j += 8)
        *(v8bf*)&sW[ib ^ 1][tid * 32 + j] = *(const v8bf*)(src + j);
#endif
    }
    __builtin_prefetch(Ctx + (size_t)waveRow * D_MODEL + kk + 128, 0, 0);
    v16bf a0 = load_a_tile(Ctx + (size_t)waveRow * D_MODEL + kk, D_MODEL);
    v16bf a1 = load_a_tile(Ctx + (size_t)(waveRow + 16) * D_MODEL + kk, D_MODEL);
    const bf16_t* sWb = &sW[ib][0];
#pragma unroll
    for (int ni = 0; ni < 4; ++ni) {
      v16bf b = load_b_tile(sWb + (size_t)(localN + ni * 16) * 32, 32);
      acc[0][ni] = wmma_bf16(a0, b, acc[0][ni]);
      acc[1][ni] = wmma_bf16(a1, b, acc[1][ni]);
    }
  }

  const int hi8 = (lane >> 4) << 3;
  const int nl = lane & 15;
#pragma unroll
  for (int mi = 0; mi < 2; ++mi)
#pragma unroll
    for (int v = 0; v < 8; ++v) {
      int t = waveRow + mi * 16 + hi8 + v;
#pragma unroll
      for (int ni = 0; ni < 4; ++ni)
        Out[(size_t)t * D_MODEL + waveCol + ni * 16 + nl] = acc[mi][ni][v];
    }
}

// ---------- launcher ----------

extern "C" void kernel_launch(void* const* d_in, const int* in_sizes, int n_in,
                              void* d_out, int out_size, void* d_ws, size_t ws_size,
                              hipStream_t stream) {
  (void)in_sizes; (void)n_in; (void)out_size; (void)ws_size;
  const float* hs   = (const float*)d_in[0];
  const float* Wqkv = (const float*)d_in[1];
  const float* Wo   = (const float*)d_in[2];
  const int*   pid  = (const int*)d_in[3];
  float* out = (float*)d_out;

  char* ws = (char*)d_ws;
  size_t off = 0;
  bf16_t* hsb   = (bf16_t*)(ws + off); off += (size_t)T_TOK * D_MODEL * 2;
  bf16_t* wqkvT = (bf16_t*)(ws + off); off += (size_t)N_QKV * D_MODEL * 2;
  bf16_t* woT   = (bf16_t*)(ws + off); off += (size_t)D_MODEL * D_MODEL * 2;
  bf16_t* Qb    = (bf16_t*)(ws + off); off += (size_t)NH * T_TOK * DHD * 2;
  bf16_t* Kb    = (bf16_t*)(ws + off); off += (size_t)NH * T_TOK * DHD * 2;
  bf16_t* Vt    = (bf16_t*)(ws + off); off += (size_t)NH * T_TOK * DHD * 2;
  bf16_t* Ctx   = (bf16_t*)(ws + off); off += (size_t)T_TOK * D_MODEL * 2;

  int n1 = T_TOK * D_MODEL;
  cvt_f32_bf16<<<(n1 + 255) / 256, 256, 0, stream>>>(hs, hsb, n1);
  int n2 = D_MODEL * N_QKV;
  cvt_transpose<<<(n2 + 255) / 256, 256, 0, stream>>>(Wqkv, wqkvT, D_MODEL, N_QKV);
  int n3 = D_MODEL * D_MODEL;
  cvt_transpose<<<(n3 + 255) / 256, 256, 0, stream>>>(Wo, woT, D_MODEL, D_MODEL);

  qkv_rope_kernel<<<576, 256, 0, stream>>>(hsb, wqkvT, pid, Qb, Kb, Vt);
  attn_kernel<<<384, 256, 0, stream>>>(Qb, Kb, Vt, Ctx);
  out_proj_kernel<<<192, 256, 0, stream>>>(Ctx, woT, out);
}
